// MultiQueryAttention_63196148793904
// MI455X (gfx1250) — matrix operand order repaired and checked by reading, not compiled, not run on hardware
//
#include <hip/hip_runtime.h>
#include <hip/hip_fp16.h>
#include <math.h>

typedef __attribute__((ext_vector_type(16))) _Float16 v16h;
typedef __attribute__((ext_vector_type(8)))  float    v8f;

union FragA { v16h h; uint4 q[2]; };

#define WMMA_F32_F16(a, b, c) \
  __builtin_amdgcn_wmma_f32_16x16x32_f16(false, (a), false, (b), (short)0, (c), false, false)

__device__ inline unsigned pk2h(float lo, float hi) {
  union { unsigned u; _Float16 h[2]; } z;
  z.h[0] = (_Float16)lo;
  z.h[1] = (_Float16)hi;
  return z.u;
}

// Async 16B global->LDS copy (CDNA5 GLOBAL_LOAD_ASYNC_TO_LDS_B128, ASYNCcnt).
// VDST operand = wave-relative LDS byte offset (low 32 bits of generic LDS ptr).
__device__ inline void async_ld16(void* lds, const void* g) {
  unsigned l = (unsigned)(uintptr_t)lds;
  unsigned long long a = (unsigned long long)(uintptr_t)g;
  asm volatile("global_load_async_to_lds_b128 %0, %1, off" ::"v"(l), "v"(a) : "memory");
}
__device__ inline void async_wait0() {
  asm volatile("s_wait_asynccnt 0x0" ::: "memory");
}

// ---------------------------------------------------------------------------
// Tiled GEMM: C[M,N] = A[M,K] * W[K,N]; A is f32 or f16, W is f32, C is f16/f32.
// Block = 256 threads (8 waves), tile 128(M) x 64(N), K-step 32.
// Waves arranged 4x2, each wave computes a 32x32 tile via 2x2 WMMA fragments.
// ---------------------------------------------------------------------------
template <typename AT, typename CT>
__global__ __launch_bounds__(256) void gemm_f16wmma(const AT* __restrict__ A,
                                                    const float* __restrict__ W,
                                                    CT* __restrict__ C,
                                                    int M, int N, int K) {
  __shared__ __align__(16) _Float16 As[128][40];  // stride 40 halves (80B, 16B-aligned)
  __shared__ __align__(16) _Float16 Wt[64][40];   // W tile transposed [N][K]

  const int t = threadIdx.x;
  const int lane = t & 31;
  const int wave = t >> 5;
  const int half16 = lane >> 4;
  const int l16 = lane & 15;
  const int m0 = blockIdx.y * 128;
  const int n0 = blockIdx.x * 64;
  const int wm = (wave >> 1) * 32;
  const int wn = (wave & 1) * 32;

  v8f acc[2][2] = {};

  for (int k0 = 0; k0 < K; k0 += 32) {
    // ---- Stage A tile 128x32 into LDS as f16 ----
    if constexpr (sizeof(AT) == 2) {
      // f16 source: async 16B copies straight to LDS (no VGPR round trip)
#pragma unroll
      for (int i = 0; i < 2; ++i) {
        int idx = i * 256 + t;
        int r = idx >> 2, c = (idx & 3) * 8;
        async_ld16(&As[r][c], &A[(size_t)(m0 + r) * K + k0 + c]);
      }
    } else {
      // f32 source: float4 -> 4 halves (b64 store), 4 iters
#pragma unroll
      for (int i = 0; i < 4; ++i) {
        int idx = i * 256 + t;
        int r = idx >> 3, c = (idx & 7) * 4;
        const float4 d = *reinterpret_cast<const float4*>(&A[(size_t)(m0 + r) * K + k0 + c]);
        union { uint2 u; _Float16 h[4]; } z;
        z.h[0] = (_Float16)d.x; z.h[1] = (_Float16)d.y;
        z.h[2] = (_Float16)d.z; z.h[3] = (_Float16)d.w;
        *reinterpret_cast<uint2*>(&As[r][c]) = z.u;
      }
    }
    // ---- Stage W tile transposed 64(N) x 32(K), float4 loads ----
#pragma unroll
    for (int i = 0; i < 2; ++i) {
      int idx = i * 256 + t;
      int kk = idx >> 4, n = (idx & 15) * 4;
      const float4 d = *reinterpret_cast<const float4*>(&W[(size_t)(k0 + kk) * N + n0 + n]);
      Wt[n + 0][kk] = (_Float16)d.x;
      Wt[n + 1][kk] = (_Float16)d.y;
      Wt[n + 2][kk] = (_Float16)d.z;
      Wt[n + 3][kk] = (_Float16)d.w;
    }
    if (k0 + 32 < K) {
      __builtin_prefetch(&A[(size_t)(m0 + (t >> 1)) * K + k0 + 32 + (t & 1) * 16], 0, 0);
      __builtin_prefetch(&W[(size_t)(k0 + 32 + (t >> 3)) * N + n0 + (t & 7) * 8], 0, 0);
    }
    if constexpr (sizeof(AT) == 2) async_wait0();
    __syncthreads();

    FragA a[2], b[2];
#pragma unroll
    for (int fm = 0; fm < 2; ++fm) {
      int row = wm + fm * 16 + l16;
      int kb = half16 * 8;  // A layout: halves 0-7 -> K[kb,kb+8), 8-15 -> K[kb+16,kb+24)
      a[fm].q[0] = *reinterpret_cast<const uint4*>(&As[row][kb]);
      a[fm].q[1] = *reinterpret_cast<const uint4*>(&As[row][kb + 16]);
    }
#pragma unroll
    for (int fn = 0; fn < 2; ++fn) {
      int col = wn + fn * 16 + l16;
      int kb = half16 * 8;  // B layout: halves 0-7 -> K[kb,kb+8), 8-15 -> K[kb+16,kb+24)
      b[fn].q[0] = *reinterpret_cast<const uint4*>(&Wt[col][kb]);
      b[fn].q[1] = *reinterpret_cast<const uint4*>(&Wt[col][kb + 16]);
    }
#pragma unroll
    for (int fm = 0; fm < 2; ++fm)
#pragma unroll
      for (int fn = 0; fn < 2; ++fn)
        acc[fm][fn] = WMMA_F32_F16(a[fm].h, b[fn].h, acc[fm][fn]);

    __syncthreads();
  }

  // C/D layout: VGPR r -> M = r (lanes 0-15) / r+8 (lanes 16-31), N = lane%16
#pragma unroll
  for (int fm = 0; fm < 2; ++fm)
#pragma unroll
    for (int fn = 0; fn < 2; ++fn) {
      int col = n0 + wn + fn * 16 + l16;
      int rowb = m0 + wm + fm * 16 + half16 * 8;
#pragma unroll
      for (int r = 0; r < 8; ++r)
        C[(size_t)(rowb + r) * N + col] = (CT)acc[fm][fn][r];
    }
}

// ---------------------------------------------------------------------------
// Flash-attention, transposed orientation (multi-query: shared KV head).
// Grid: (S/128, H, B). Block = 256 threads; wave w owns q rows [q0, q0+16).
//   S^T = K * Q^T   (A = K tile from LDS, B = Q frags from global, resident)
//   softmax: per-lane stats (q = lane%16), one shfl_xor(16) per reduction
//   O^T  += V^T * P^T (A = V^T tile from LDS, B = P^T built in-register
//                      with 8 packed shfl_xor(16) exchanges)
// K tile staged with async global->LDS b128; V staged manually (transpose).
// ---------------------------------------------------------------------------
__global__ __launch_bounds__(256) void mqa_attention(const _Float16* __restrict__ Q,
                                                     const _Float16* __restrict__ Kb,
                                                     const _Float16* __restrict__ Vb,
                                                     _Float16* __restrict__ O,
                                                     float scale) {
  __shared__ __align__(16) _Float16 Kt[32][72];  // K tile [s][d], 144B stride (16B-aligned)
  __shared__ __align__(16) _Float16 Vt[64][40];  // V tile transposed [d][s], 80B stride

  const int t = threadIdx.x;
  const int lane = t & 31;
  const int wave = t >> 5;
  const int half16 = lane >> 4;
  const int l16 = lane & 15;

  const int qt = blockIdx.x;
  const int h = blockIdx.y;
  const int b = blockIdx.z;
  const size_t sbase = (size_t)b * 2048;
  const int q0 = qt * 128 + wave * 16;

  // Q^T B-fragments (d 0..31 and 32..63): lane -> n = q = l16, K = d in table order (8h, 8h+16)
  FragA qf[2];
  {
    const _Float16* qp = Q + (sbase + q0 + l16) * 1024 + h * 64 + half16 * 8;
#pragma unroll
    for (int f = 0; f < 2; ++f) {
      qf[f].q[0] = *reinterpret_cast<const uint4*>(qp + f * 32);
      qf[f].q[1] = *reinterpret_cast<const uint4*>(qp + f * 32 + 16);
    }
  }

  v8f acc[4] = {};  // O^T accumulators: M = d (4 groups of 16), N = q = l16
  float mrun = -1e30f, lrun = 0.f;

  for (int kv = 0; kv < 2048; kv += 32) {
    // ---- Staging: K async (16B/lane), V manual transpose ----
    {
      int row = t >> 3;      // 0..31
      int c0 = (t & 7) * 8;  // 0..56
      async_ld16(&Kt[row][c0], Kb + (sbase + kv + row) * 64 + c0);
      const uint4 vd = *reinterpret_cast<const uint4*>(Vb + (sbase + kv + row) * 64 + c0);
      const _Float16* vh = reinterpret_cast<const _Float16*>(&vd);
#pragma unroll
      for (int j = 0; j < 8; ++j) Vt[c0 + j][row] = vh[j];
      if (kv + 32 < 2048) {
        __builtin_prefetch(Kb + (sbase + kv + 32 + row) * 64 + c0, 0, 0);
        __builtin_prefetch(Vb + (sbase + kv + 32 + row) * 64 + c0, 0, 0);
      }
    }
    async_wait0();
    __syncthreads();

    // ---- S^T[32 kv x 16 q]: two 16-row kv groups, K-dim (d) split 2x32 ----
    v8f st[2];
#pragma unroll
    for (int cg = 0; cg < 2; ++cg) {
      FragA k0f, k1f;
      int row = cg * 16 + l16;  // A-frag: M = kv row
      int kb = half16 * 8;      // A layout within d-block
      k0f.q[0] = *reinterpret_cast<const uint4*>(&Kt[row][kb]);
      k0f.q[1] = *reinterpret_cast<const uint4*>(&Kt[row][kb + 16]);
      k1f.q[0] = *reinterpret_cast<const uint4*>(&Kt[row][32 + kb]);
      k1f.q[1] = *reinterpret_cast<const uint4*>(&Kt[row][32 + kb + 16]);
      v8f s = {};
      s = WMMA_F32_F16(k0f.h, qf[0].h, s);
      s = WMMA_F32_F16(k1f.h, qf[1].h, s);
      st[cg] = s;
    }

    // ---- Online softmax: per-lane stats (q = l16), kv split across halves
    float vmax = -1e30f;
#pragma unroll
    for (int cg = 0; cg < 2; ++cg)
#pragma unroll
      for (int r = 0; r < 8; ++r) {
        st[cg][r] *= scale;
        vmax = fmaxf(vmax, st[cg][r]);
      }
    vmax = fmaxf(vmax, __shfl_xor(vmax, 16));
    const float mnew = fmaxf(mrun, vmax);
    const float corr = __expf(mrun - mnew);
    mrun = mnew;

    float p0[8], p1[8];
    float rsum = 0.f;
#pragma unroll
    for (int r = 0; r < 8; ++r) {
      p0[r] = __expf(st[0][r] - mnew);
      p1[r] = __expf(st[1][r] - mnew);
      rsum += p0[r] + p1[r];
    }
    rsum += __shfl_xor(rsum, 16);
    lrun = lrun * corr + rsum;
#pragma unroll
    for (int g = 0; g < 4; ++g)
#pragma unroll
      for (int r = 0; r < 8; ++r) acc[g][r] *= corr;

    // ---- Build P^T B-fragment in-register (K = kv, N = q = l16) ----
    unsigned pa[4], pb[4];
#pragma unroll
    for (int i = 0; i < 4; ++i) {
      pa[i] = pk2h(p0[2 * i], p0[2 * i + 1]);
      pb[i] = pk2h(p1[2 * i], p1[2 * i + 1]);
    }
    unsigned lo[4], hi[4];
#pragma unroll
    for (int i = 0; i < 4; ++i) {
      unsigned xa = (unsigned)__shfl_xor((int)pa[i], 16);
      unsigned xb = (unsigned)__shfl_xor((int)pb[i], 16);
      lo[i] = half16 ? xb : pa[i];
      hi[i] = half16 ? pb[i] : xa;
    }
    FragA pf;
    pf.q[0] = make_uint4(lo[0], lo[1], lo[2], lo[3]);
    pf.q[1] = make_uint4(hi[0], hi[1], hi[2], hi[3]);

    // ---- O^T += V^T * P^T : 4 d-groups of 16 rows ----
#pragma unroll
    for (int g = 0; g < 4; ++g) {
      FragA vf;
      int d = g * 16 + l16;  // A-frag: M = d row of Vt
      int kb = half16 * 16;
      vf.q[0] = *reinterpret_cast<const uint4*>(&Vt[d][kb]);
      vf.q[1] = *reinterpret_cast<const uint4*>(&Vt[d][kb + 8]);
      acc[g] = WMMA_F32_F16(vf.h, pf.h, acc[g]);
    }
    __syncthreads();
  }

  // ---- Normalize & write: O^T layout -> row q = l16, d contiguous along VGPRs
  const float linv = 1.0f / lrun;
  _Float16* op = O + (sbase + q0 + l16) * 1024 + h * 64 + half16 * 8;
#pragma unroll
  for (int g = 0; g < 4; ++g) {
    unsigned w[4];
#pragma unroll
    for (int i = 0; i < 4; ++i)
      w[i] = pk2h(acc[g][2 * i] * linv, acc[g][2 * i + 1] * linv);
    *reinterpret_cast<uint4*>(op + g * 16) = make_uint4(w[0], w[1], w[2], w[3]);
  }
}

// ---------------------------------------------------------------------------
extern "C" void kernel_launch(void* const* d_in, const int* in_sizes, int n_in,
                              void* d_out, int out_size, void* d_ws, size_t ws_size,
                              hipStream_t stream) {
  const float* hs = (const float*)d_in[0];
  const float* Wq = (const float*)d_in[1];
  const float* Wk = (const float*)d_in[2];
  const float* Wv = (const float*)d_in[3];
  const float* Wo = (const float*)d_in[4];
  float* out = (float*)d_out;

  const int M = 2 * 2048;  // B*S
  const int HID = 1024, D = 64;

  _Float16* Qb = (_Float16*)d_ws;       // [M, 1024] f16
  _Float16* Kb = Qb + (size_t)M * HID;  // [M, 64]   f16
  _Float16* Vb = Kb + (size_t)M * D;    // [M, 64]   f16
  _Float16* AO = Vb + (size_t)M * D;    // [M, 1024] f16

  dim3 blk(256);
  gemm_f16wmma<float, _Float16><<<dim3(HID / 64, M / 128), blk, 0, stream>>>(hs, Wq, Qb, M, HID, HID);
  gemm_f16wmma<float, _Float16><<<dim3(D / 64, M / 128), blk, 0, stream>>>(hs, Wk, Kb, M, D, HID);
  gemm_f16wmma<float, _Float16><<<dim3(D / 64, M / 128), blk, 0, stream>>>(hs, Wv, Vb, M, D, HID);
  mqa_attention<<<dim3(16, 16, 2), blk, 0, stream>>>(Qb, Kb, Vb, AO, 0.125f);
  gemm_f16wmma<_Float16, float><<<dim3(HID / 64, M / 128), blk, 0, stream>>>(AO, Wo, out, M, HID, HID);
}
